// MyRNN_10737418240149
// MI455X (gfx1250) — matrix operand order repaired and checked by reading, not compiled, not run on hardware
//
#include <hip/hip_runtime.h>
#include <math.h>

// Problem dims (fixed by reference)
#define BB 8
#define TT 128
#define FF 512
#define HH 512
#define CC 128
#define MM (BB * TT)   // 1024 independent rows

typedef __attribute__((ext_vector_type(16))) __bf16 v16bf;
typedef __attribute__((ext_vector_type(8)))  float  v8f;

union BFrag { v16bf v; uint4 q[2]; };

static __device__ __forceinline__ unsigned short f2bf(float f) {
    // round-to-nearest-even f32 -> bf16 (bits)
    unsigned int u = __float_as_uint(f);
    unsigned int r = (u + 0x7FFFu + ((u >> 16) & 1u)) >> 16;
    return (unsigned short)r;
}

// ---------------------------------------------------------------------------
// Kernel 1: convert x, W1 (split h|x), W2 to bf16
// ---------------------------------------------------------------------------
__global__ void convert_kernel(const float* __restrict__ x,
                               const float* __restrict__ W1,
                               const float* __restrict__ W2,
                               unsigned short* __restrict__ x_bf,
                               unsigned short* __restrict__ W1x_bf,
                               unsigned short* __restrict__ W1h_bf,
                               unsigned short* __restrict__ W2_bf) {
    const int N0 = MM * FF;          // x
    const int N1 = HH * (HH + FF);   // W1
    const int N2 = CC * HH;          // W2
    int i = blockIdx.x * blockDim.x + threadIdx.x;
    if (i < N0) {
        x_bf[i] = f2bf(x[i]);
    } else if (i < N0 + N1) {
        int j = i - N0;
        int h = j / (HH + FF);
        int c = j - h * (HH + FF);
        unsigned short v = f2bf(W1[j]);
        if (c < HH) W1h_bf[h * HH + c] = v;
        else        W1x_bf[h * FF + (c - HH)] = v;
    } else if (i < N0 + N1 + N2) {
        int j = i - N0 - N1;
        W2_bf[j] = f2bf(W2[j]);
    }
}

// ---------------------------------------------------------------------------
// Kernel 2: C[MxN] = A[MxK] * B[NxK]^T + bias[N]   (bf16 in, f32 out)
// one wave per 16x16 output tile, v_wmma_f32_16x16x32_bf16
// ---------------------------------------------------------------------------
__global__ __launch_bounds__(256) void gemm_nt_bias(
        const unsigned short* __restrict__ A,
        const unsigned short* __restrict__ Bm,
        const float* __restrict__ bias,
        float* __restrict__ C,
        int M, int N, int K) {
    int wave = (blockIdx.x * blockDim.x + threadIdx.x) >> 5;
    int lane = threadIdx.x & 31;
    int tN = N >> 4;
    int tiles = (M >> 4) * tN;
    if (wave >= tiles) return;            // wave-uniform: EXEC stays all-ones
    int m0 = (wave / tN) << 4;
    int n0 = (wave % tN) << 4;
    int row = lane & 15;
    int hi  = lane >> 4;

    v8f acc;
    float bv = bias[n0 + row];
    #pragma unroll
    for (int j = 0; j < 8; ++j) acc[j] = bv;

    const unsigned short* arow = A  + (size_t)(m0 + row) * K;
    const unsigned short* brow = Bm + (size_t)(n0 + row) * K;

    for (int kk = 0; kk < K; kk += 32) {
        BFrag a, b;
        a.q[0] = *(const uint4*)(arow + kk + hi * 8);        // K = hi*8 .. +7
        a.q[1] = *(const uint4*)(arow + kk + 16 + hi * 8);   // K = 16+hi*8 ..
        b.q[0] = *(const uint4*)(brow + kk + hi * 8);        // K = hi*8 .. +7
        b.q[1] = *(const uint4*)(brow + kk + 16 + hi * 8);
        acc = __builtin_amdgcn_wmma_f32_16x16x32_bf16(
                  false, a.v, false, b.v, (short)0, acc, false, false);
    }
    #pragma unroll
    for (int j = 0; j < 8; ++j)
        C[(size_t)(m0 + hi * 8 + j) * N + n0 + row] = acc[j];
}

// ---------------------------------------------------------------------------
// Kernel 3: 128x fixed-point iteration  h = tanh(h * W1h^T + xa)
// 64 blocks x 1024 threads (32 waves); block owns 16 rows; each wave owns one
// 16-column tile for the whole kernel. B fragments for K in [0,256) are held
// in registers across all iterations (their data never changes); K in
// [256,512) streams from the L2-resident W1h.
// ---------------------------------------------------------------------------
#define HPAD 520   // 16B-aligned padded row stride (elements) to dodge bank conflicts

__global__ __launch_bounds__(1024) void rnn_iter_kernel(
        const unsigned short* __restrict__ W1h,   // H x H bf16 (L2-resident)
        const float* __restrict__ xa,             // M x H f32
        float* __restrict__ h_f32,                // M x H f32 (second d_out region)
        unsigned short* __restrict__ h_bf) {      // M x H bf16 (workspace)
    __shared__ unsigned short hbuf[16 * HPAD];
    const int tid  = threadIdx.x;
    const int wave = tid >> 5;            // 0..31  -> n-tile index
    const int lane = tid & 31;
    const int row  = lane & 15;
    const int hi   = lane >> 4;
    const int m0   = blockIdx.x * 16;
    const int n    = (wave << 4) + row;   // this lane's output column

    // h0 = 0
    for (int i = tid; i < 16 * HPAD; i += blockDim.x) hbuf[i] = 0;

    // xa fragment stays in registers for all 128 iterations
    v8f xfrag;
    #pragma unroll
    for (int j = 0; j < 8; ++j)
        xfrag[j] = xa[(size_t)(m0 + hi * 8 + j) * HH + n];

    // register-resident B fragments for K in [0,256): 8 k-steps x 32B/lane
    const unsigned short* brow = W1h + (size_t)n * HH;
    uint4 breg[16];
    #pragma unroll
    for (int ks = 0; ks < 8; ++ks) {
        breg[2 * ks + 0] = *(const uint4*)(brow + ks * 32 + hi * 8);
        breg[2 * ks + 1] = *(const uint4*)(brow + ks * 32 + 16 + hi * 8);
    }
    __syncthreads();

    v8f acc;
    for (int it = 0; it < TT; ++it) {
        acc = xfrag;
        const unsigned short* ar = hbuf + row * HPAD;

        // K in [0,256): A from LDS, B from registers -> pure ds + wmma
        #pragma unroll
        for (int ks = 0; ks < 8; ++ks) {
            BFrag a, b;
            a.q[0] = *(const uint4*)(ar + ks * 32 + hi * 8);
            a.q[1] = *(const uint4*)(ar + ks * 32 + 16 + hi * 8);
            b.q[0] = breg[2 * ks + 0];
            b.q[1] = breg[2 * ks + 1];
            acc = __builtin_amdgcn_wmma_f32_16x16x32_bf16(
                      false, a.v, false, b.v, (short)0, acc, false, false);
        }
        // K in [256,512): B streamed from L2
        #pragma unroll
        for (int ks = 8; ks < 16; ++ks) {
            BFrag a, b;
            a.q[0] = *(const uint4*)(ar + ks * 32 + hi * 8);
            a.q[1] = *(const uint4*)(ar + ks * 32 + 16 + hi * 8);
            b.q[0] = *(const uint4*)(brow + ks * 32 + hi * 8);
            b.q[1] = *(const uint4*)(brow + ks * 32 + 16 + hi * 8);
            acc = __builtin_amdgcn_wmma_f32_16x16x32_bf16(
                      false, a.v, false, b.v, (short)0, acc, false, false);
        }
        #pragma unroll
        for (int j = 0; j < 8; ++j) acc[j] = tanhf(acc[j]);

        __syncthreads();   // all reads of hbuf for this iteration done
        #pragma unroll
        for (int j = 0; j < 8; ++j)
            hbuf[(hi * 8 + j) * HPAD + n] = f2bf(acc[j]);
        __syncthreads();   // new h visible to all waves
    }

    // final h: f32 (exact tanh outputs) + bf16 (for the logits GEMM)
    #pragma unroll
    for (int j = 0; j < 8; ++j) {
        int m = m0 + hi * 8 + j;
        h_f32[(size_t)m * HH + n] = acc[j];
        h_bf [(size_t)m * HH + n] = f2bf(acc[j]);
    }
}

// ---------------------------------------------------------------------------
// Kernel 4: logits = h * W2^T + b2, softmax over C=128
// 64 blocks x 256 threads (8 waves); wave w -> cols [16w,16w+16)
// ---------------------------------------------------------------------------
#define LPAD 132

__global__ __launch_bounds__(256) void logits_softmax_kernel(
        const unsigned short* __restrict__ h_bf,   // M x H bf16
        const unsigned short* __restrict__ W2bf,   // C x H bf16
        const float* __restrict__ b2,              // C
        float* __restrict__ out) {                 // M x C
    __shared__ float lbuf[16 * LPAD];
    const int tid  = threadIdx.x;
    const int wave = tid >> 5;       // 0..7
    const int lane = tid & 31;
    const int row  = lane & 15;
    const int hi   = lane >> 4;
    const int m0   = blockIdx.x * 16;
    const int n0   = wave * 16;

    v8f acc;
    float bv = b2[n0 + row];
    #pragma unroll
    for (int j = 0; j < 8; ++j) acc[j] = bv;

    const unsigned short* arow = h_bf + (size_t)(m0 + row) * HH;
    const unsigned short* brow = W2bf + (size_t)(n0 + row) * HH;
    for (int kk = 0; kk < HH; kk += 32) {
        BFrag a, b;
        a.q[0] = *(const uint4*)(arow + kk + hi * 8);
        a.q[1] = *(const uint4*)(arow + kk + 16 + hi * 8);
        b.q[0] = *(const uint4*)(brow + kk + hi * 8);
        b.q[1] = *(const uint4*)(brow + kk + 16 + hi * 8);
        acc = __builtin_amdgcn_wmma_f32_16x16x32_bf16(
                  false, a.v, false, b.v, (short)0, acc, false, false);
    }
    #pragma unroll
    for (int j = 0; j < 8; ++j)
        lbuf[(hi * 8 + j) * LPAD + n0 + row] = acc[j];
    __syncthreads();

    if (tid < 16) {                       // one thread per row; trivial work
        float mx = -3.402823466e+38f;
        for (int c = 0; c < CC; ++c) mx = fmaxf(mx, lbuf[tid * LPAD + c]);
        float s = 0.0f;
        for (int c = 0; c < CC; ++c) {
            float e = __expf(lbuf[tid * LPAD + c] - mx);
            lbuf[tid * LPAD + c] = e;
            s += e;
        }
        float inv = 1.0f / s;
        for (int c = 0; c < CC; ++c)
            out[(size_t)(m0 + tid) * CC + c] = lbuf[tid * LPAD + c] * inv;
    }
}

// ---------------------------------------------------------------------------
// Launch
// ---------------------------------------------------------------------------
extern "C" void kernel_launch(void* const* d_in, const int* in_sizes, int n_in,
                              void* d_out, int out_size, void* d_ws, size_t ws_size,
                              hipStream_t stream) {
    (void)in_sizes; (void)n_in; (void)out_size; (void)ws_size;
    const float* x  = (const float*)d_in[0];   // (B,T,F)
    const float* W1 = (const float*)d_in[1];   // (H, H+F)
    const float* b1 = (const float*)d_in[2];   // (H)
    const float* W2 = (const float*)d_in[3];   // (C, H)
    const float* b2 = (const float*)d_in[4];   // (C)

    // workspace layout (256B-aligned offsets)
    char* ws = (char*)d_ws;
    unsigned short* x_bf   = (unsigned short*)(ws + 0);         // 1,048,576 B
    unsigned short* W1x_bf = (unsigned short*)(ws + 1048576);   //   524,288 B
    unsigned short* W1h_bf = (unsigned short*)(ws + 1572864);   //   524,288 B
    unsigned short* W2_bf  = (unsigned short*)(ws + 2097152);   //   131,072 B
    float*          xa     = (float*)         (ws + 2228224);   // 2,097,152 B
    unsigned short* h_bf   = (unsigned short*)(ws + 4325376);   // 1,048,576 B

    float* out_sm = (float*)d_out;              // (B,T,C) softmax
    float* out_h  = out_sm + (size_t)MM * CC;   // (B,T,H) final h (f32)

    // 1) bf16 conversion
    {
        int total = MM * FF + HH * (HH + FF) + CC * HH;
        int blocks = (total + 255) / 256;
        convert_kernel<<<blocks, 256, 0, stream>>>(x, W1, W2,
                                                   x_bf, W1x_bf, W1h_bf, W2_bf);
    }
    // 2) xa = x * W1x^T + b1   (1024 x 512, K=512) -> 2048 tiles -> 2048 waves
    {
        int tiles = (MM / 16) * (HH / 16);
        int blocks = (tiles * 32 + 255) / 256;
        gemm_nt_bias<<<blocks, 256, 0, stream>>>(x_bf, W1x_bf, b1, xa,
                                                 MM, HH, FF);
    }
    // 3) 128 fixed-point iterations (independent per 16-row block)
    rnn_iter_kernel<<<MM / 16, 1024, 0, stream>>>(W1h_bf, xa, out_h, h_bf);

    // 4) logits + softmax
    logits_softmax_kernel<<<MM / 16, 256, 0, stream>>>(h_bf, W2_bf, b2, out_sm);
}
